// Similarity_71167608095276
// MI455X (gfx1250) — matrix operand order repaired and checked by reading, not compiled, not run on hardware
//
#include <hip/hip_runtime.h>

#define N_ROWS 8192
#define M_COLS 1024
#define D_DIM  1024
#define LDSK   40   // padded LDS row stride (bf16 elems) for a 32-wide K tile

typedef __bf16 v16bf __attribute__((ext_vector_type(16)));
typedef float  v8f   __attribute__((ext_vector_type(8)));

union Frag {
  v16bf v;
  uint4 q[2];
};

// ---- fp32 -> bf16 (RTNE) helpers via bit manipulation ----
__device__ __forceinline__ unsigned int f2bf_bits(float f) {
  unsigned int x = __float_as_uint(f);
  return (x + 0x7FFFu + ((x >> 16) & 1u)) >> 16;
}
__device__ __forceinline__ float bf_to_f(unsigned int b) {
  return __uint_as_float(b << 16);
}
// x ~= hi + lo with hi,lo bf16; captures ~16-18 mantissa bits total
__device__ __forceinline__ void split_hi_lo(float x, unsigned int& hi, unsigned int& lo) {
  hi = f2bf_bits(x);
  lo = f2bf_bits(x - bf_to_f(hi));
}

// ---------------------------------------------------------------------------
// Pass 1 (fast path): convert h*w3 and u to bf16 hi/lo pairs in workspace,
// and fuse the row-dot bias terms a[n]=h[n].w1, b[m]=u[m].w2.
// One 256-thread block per row; each thread owns 4 columns.
// ---------------------------------------------------------------------------
__global__ __launch_bounds__(256)
void convert_kernel(const float* __restrict__ h, const float* __restrict__ u,
                    const float* __restrict__ w,
                    unsigned short* __restrict__ Ahi, unsigned short* __restrict__ Alo,
                    unsigned short* __restrict__ Bhi, unsigned short* __restrict__ Blo,
                    float* __restrict__ avec, float* __restrict__ bvec)
{
  __shared__ float red[8];
  const int tid  = threadIdx.x;
  const int lane = tid & 31;
  const int wid  = tid >> 5;
  const int row  = blockIdx.x;          // 0..9215
  const int c4   = tid * 4;
  const bool isA = (row < N_ROWS);

  const float* src;
  const float* wvec;
  unsigned short* hiP;
  unsigned short* loP;
  float* dout;
  int r;
  if (isA) {
    r = row;          src = h + (size_t)r * D_DIM; wvec = w;
    hiP = Ahi; loP = Alo; dout = avec + r;
  } else {
    r = row - N_ROWS; src = u + (size_t)r * D_DIM; wvec = w + D_DIM;
    hiP = Bhi; loP = Blo; dout = bvec + r;
  }

  const float4 x = *(const float4*)&src[c4];
  const float4 y = *(const float4*)&wvec[c4];
  float dot = x.x * y.x + x.y * y.y + x.z * y.z + x.w * y.w;

  float4 v = x;
  if (isA) {
    const float4 s3 = *(const float4*)&w[2 * D_DIM + c4];
    v.x *= s3.x; v.y *= s3.y; v.z *= s3.z; v.w *= s3.w;
  }
  unsigned int h0, h1, h2, h3, l0, l1, l2, l3;
  split_hi_lo(v.x, h0, l0); split_hi_lo(v.y, h1, l1);
  split_hi_lo(v.z, h2, l2); split_hi_lo(v.w, h3, l3);
  const size_t off = (size_t)r * D_DIM + c4;
  *(uint2*)&hiP[off] = make_uint2(h0 | (h1 << 16), h2 | (h3 << 16));
  *(uint2*)&loP[off] = make_uint2(l0 | (l1 << 16), l2 | (l3 << 16));

#pragma unroll
  for (int offx = 16; offx > 0; offx >>= 1) dot += __shfl_xor(dot, offx, 32);
  if (lane == 0) red[wid] = dot;
  __syncthreads();
  if (tid == 0) {
    float s = 0.0f;
#pragma unroll
    for (int i = 0; i < 8; ++i) s += red[i];
    *dout = s;
  }
}

// ---- shared WMMA tile-compute over one staged 128x32 K-slab ----
__device__ __forceinline__ void wmma_block(
    const unsigned short (*lAhi)[LDSK], const unsigned short (*lAlo)[LDSK],
    const unsigned short (*lBhi)[LDSK], const unsigned short (*lBlo)[LDSK],
    int waveM, int waveN, int lrow, int half, v8f acc[2][4])
{
  // A fragments (lanes 0-15 K 0..15; lanes 16-31 K 16..31)
  Frag ahi[2], alo[2];
#pragma unroll
  for (int i = 0; i < 2; ++i) {
    const int r  = waveM + i * 16 + lrow;
    const int kb = half * 16;
    ahi[i].q[0] = *(const uint4*)&lAhi[r][kb];
    ahi[i].q[1] = *(const uint4*)&lAhi[r][kb + 8];
    alo[i].q[0] = *(const uint4*)&lAlo[r][kb];
    alo[i].q[1] = *(const uint4*)&lAlo[r][kb + 8];
  }
  // B fragments (lanes 0-15 K 0..15; lanes 16-31 K 16..31)
#pragma unroll
  for (int j = 0; j < 4; ++j) {
    const int n  = waveN + j * 16 + lrow;
    const int kb = half * 16;
    Frag bhi, blo;
    bhi.q[0] = *(const uint4*)&lBhi[n][kb];
    bhi.q[1] = *(const uint4*)&lBhi[n][kb + 8];
    blo.q[0] = *(const uint4*)&lBlo[n][kb];
    blo.q[1] = *(const uint4*)&lBlo[n][kb + 8];
#pragma unroll
    for (int i = 0; i < 2; ++i) {
      acc[i][j] = __builtin_amdgcn_wmma_f32_16x16x32_bf16(
          false, ahi[i].v, false, bhi.v, (short)0, acc[i][j], false, false);
      acc[i][j] = __builtin_amdgcn_wmma_f32_16x16x32_bf16(
          false, ahi[i].v, false, blo.v, (short)0, acc[i][j], false, false);
      acc[i][j] = __builtin_amdgcn_wmma_f32_16x16x32_bf16(
          false, alo[i].v, false, bhi.v, (short)0, acc[i][j], false, false);
    }
  }
}

// ---------------------------------------------------------------------------
// Pass 2 (fast path): GEMM on pre-converted bf16 hi/lo with double-buffered
// LDS (80 KB of gfx1250's 320 KB/WGP) and register-prefetched global loads.
// Global prefetch uses fixed 32-bit per-lane offsets + scalar base pointers
// advanced per K-step (SALU) so no 64-bit VALU address math in the hot loop.
// ---------------------------------------------------------------------------
__global__ __launch_bounds__(256)
void sim_gemm_pre_kernel(const unsigned short* __restrict__ Ahi,
                         const unsigned short* __restrict__ Alo,
                         const unsigned short* __restrict__ Bhi,
                         const unsigned short* __restrict__ Blo,
                         const float* __restrict__ avec,
                         const float* __restrict__ bvec,
                         float* __restrict__ out)
{
  __shared__ __align__(16) unsigned short lA[2][2][128][LDSK]; // [buf][hi/lo]
  __shared__ __align__(16) unsigned short lB[2][2][128][LDSK];

  const int tid   = threadIdx.x;
  const int lane  = tid & 31;
  const int wid   = tid >> 5;
  const int lrow  = lane & 15;
  const int half  = lane >> 4;
  const int waveM = (wid & 3) * 32;
  const int waveN = (wid >> 2) * 64;

  const int rowBaseA = blockIdx.y * 128;
  const int rowBaseB = blockIdx.x * 128;

  v8f acc[2][4];
#pragma unroll
  for (int i = 0; i < 2; ++i)
#pragma unroll
    for (int j = 0; j < 4; ++j)
#pragma unroll
      for (int e = 0; e < 8; ++e) acc[i][j][e] = 0.0f;

  // fixed per-lane offsets (element units, < 2^24) and LDS slot indices
  unsigned int offA[2], offB[2];
  int rr[2], c8[2];
#pragma unroll
  for (int i = 0; i < 2; ++i) {
    const int q = tid + 256 * i;   // uint4 index within 128x32 bf16 tile
    rr[i] = q >> 2;                // row 0..127
    c8[i] = (q & 3) * 8;           // col 0,8,16,24
    offA[i] = (unsigned int)(rowBaseA + rr[i]) * D_DIM + c8[i];
    offB[i] = (unsigned int)(rowBaseB + rr[i]) * D_DIM + c8[i];
  }

  // scalar base pointers, advanced by 32 elements per K-step
  const unsigned short* pAhi = Ahi;
  const unsigned short* pAlo = Alo;
  const unsigned short* pBhi = Bhi;
  const unsigned short* pBlo = Blo;

  uint4 rAhi[2], rAlo[2], rBhi[2], rBlo[2];

  // prologue: load K-step 0 into regs, then into buffer 0
#pragma unroll
  for (int i = 0; i < 2; ++i) {
    rAhi[i] = *(const uint4*)&pAhi[offA[i]];  rAlo[i] = *(const uint4*)&pAlo[offA[i]];
    rBhi[i] = *(const uint4*)&pBhi[offB[i]];  rBlo[i] = *(const uint4*)&pBlo[offB[i]];
  }
#pragma unroll
  for (int i = 0; i < 2; ++i) {
    *(uint4*)&lA[0][0][rr[i]][c8[i]] = rAhi[i];
    *(uint4*)&lA[0][1][rr[i]][c8[i]] = rAlo[i];
    *(uint4*)&lB[0][0][rr[i]][c8[i]] = rBhi[i];
    *(uint4*)&lB[0][1][rr[i]][c8[i]] = rBlo[i];
  }
  __syncthreads();

  for (int kt = 0; kt < D_DIM / 32; ++kt) {
    const int cur = kt & 1;
    const bool more = (kt + 1) < (D_DIM / 32);
    if (more) {
      pAhi += 32; pAlo += 32; pBhi += 32; pBlo += 32;   // SALU pointer bump
#pragma unroll
      for (int i = 0; i < 2; ++i) {
        rAhi[i] = *(const uint4*)&pAhi[offA[i]];  rAlo[i] = *(const uint4*)&pAlo[offA[i]];
        rBhi[i] = *(const uint4*)&pBhi[offB[i]];  rBlo[i] = *(const uint4*)&pBlo[offB[i]];
      }
    }

    wmma_block(lA[cur][0], lA[cur][1], lB[cur][0], lB[cur][1],
               waveM, waveN, lrow, half, acc);

    if (more) {
      const int nxt = cur ^ 1;
#pragma unroll
      for (int i = 0; i < 2; ++i) {
        *(uint4*)&lA[nxt][0][rr[i]][c8[i]] = rAhi[i];
        *(uint4*)&lA[nxt][1][rr[i]][c8[i]] = rAlo[i];
        *(uint4*)&lB[nxt][0][rr[i]][c8[i]] = rBhi[i];
        *(uint4*)&lB[nxt][1][rr[i]][c8[i]] = rBlo[i];
      }
    }
    __syncthreads();
  }

  // epilogue: + a[row] + b[col]; C layout: VGPR v -> M = v + half*8
  const int rowBase = blockIdx.y * 128 + waveM;
  const int colBase = blockIdx.x * 128 + waveN;
#pragma unroll
  for (int i = 0; i < 2; ++i) {
    const int r0 = rowBase + i * 16 + half * 8;
    float aadd[8];
#pragma unroll
    for (int v = 0; v < 8; ++v) aadd[v] = avec[r0 + v];
#pragma unroll
    for (int j = 0; j < 4; ++j) {
      const int col  = colBase + j * 16 + lrow;
      const float badd = bvec[col];
      float* op = out + (size_t)r0 * M_COLS + col;
#pragma unroll
      for (int v = 0; v < 8; ++v)
        op[(size_t)v * M_COLS] = acc[i][j][v] + aadd[v] + badd;
    }
  }
}

// ---------------------------------------------------------------------------
// Fallback path (small workspace): fused-conversion kernels.
// ---------------------------------------------------------------------------
__global__ __launch_bounds__(256)
void rowdot_kernel(const float* __restrict__ h, const float* __restrict__ u,
                   const float* __restrict__ w, float* __restrict__ avec,
                   float* __restrict__ bvec)
{
  const int lane = threadIdx.x & 31;
  const int wid  = threadIdx.x >> 5;
  const int gr   = blockIdx.x * 8 + wid;

  const float* vec;
  const float* wv;
  float* outp;
  if (gr < N_ROWS) {
    vec = h + (size_t)gr * D_DIM;  wv = w;          outp = avec + gr;
  } else {
    const int m = gr - N_ROWS;
    vec = u + (size_t)m * D_DIM;   wv = w + D_DIM;  outp = bvec + m;
  }

  float sum = 0.0f;
#pragma unroll
  for (int t = 0; t < D_DIM / (32 * 4); ++t) {
    const int o = t * 128 + lane * 4;
    const float4 x = *(const float4*)&vec[o];
    const float4 y = *(const float4*)&wv[o];
    sum += x.x * y.x + x.y * y.y + x.z * y.z + x.w * y.w;
  }
#pragma unroll
  for (int off = 16; off > 0; off >>= 1) sum += __shfl_xor(sum, off, 32);
  if (lane == 0) *outp = sum;
}

__global__ __launch_bounds__(256)
void sim_gemm_fused_kernel(const float* __restrict__ h, const float* __restrict__ u,
                           const float* __restrict__ w, const float* __restrict__ avec,
                           const float* __restrict__ bvec, float* __restrict__ out)
{
  __shared__ __align__(16) unsigned short sAhi[128][LDSK];
  __shared__ __align__(16) unsigned short sAlo[128][LDSK];
  __shared__ __align__(16) unsigned short sBhi[128][LDSK];
  __shared__ __align__(16) unsigned short sBlo[128][LDSK];
  __shared__ float w3s[D_DIM];

  const int tid   = threadIdx.x;
  const int lane  = tid & 31;
  const int wid   = tid >> 5;
  const int lrow  = lane & 15;
  const int half  = lane >> 4;
  const int waveM = (wid & 3) * 32;
  const int waveN = (wid >> 2) * 64;

  const float* hBlk = h + (size_t)blockIdx.y * 128 * D_DIM;
  const float* uBlk = u + (size_t)blockIdx.x * 128 * D_DIM;
  const float* w3   = w + 2 * D_DIM;

  for (int i = tid; i < D_DIM; i += 256) w3s[i] = w3[i];

  v8f acc[2][4];
#pragma unroll
  for (int i = 0; i < 2; ++i)
#pragma unroll
    for (int j = 0; j < 4; ++j)
#pragma unroll
      for (int e = 0; e < 8; ++e) acc[i][j][e] = 0.0f;

  for (int kt = 0; kt < D_DIM / 32; ++kt) {
    const int k0 = kt * 32;
    __syncthreads();
#pragma unroll
    for (int it = 0; it < 4; ++it) {
      const int f  = tid + it * 256;
      const int r  = f >> 3;
      const int c4 = (f & 7) * 4;
      const float4 wv = *(const float4*)&w3s[k0 + c4];
      float4 av = *(const float4*)&hBlk[(size_t)r * D_DIM + k0 + c4];
      av.x *= wv.x; av.y *= wv.y; av.z *= wv.z; av.w *= wv.w;
      unsigned int h0, h1, h2, h3, l0, l1, l2, l3;
      split_hi_lo(av.x, h0, l0); split_hi_lo(av.y, h1, l1);
      split_hi_lo(av.z, h2, l2); split_hi_lo(av.w, h3, l3);
      *(uint2*)&sAhi[r][c4] = make_uint2(h0 | (h1 << 16), h2 | (h3 << 16));
      *(uint2*)&sAlo[r][c4] = make_uint2(l0 | (l1 << 16), l2 | (l3 << 16));
      const float4 bv = *(const float4*)&uBlk[(size_t)r * D_DIM + k0 + c4];
      split_hi_lo(bv.x, h0, l0); split_hi_lo(bv.y, h1, l1);
      split_hi_lo(bv.z, h2, l2); split_hi_lo(bv.w, h3, l3);
      *(uint2*)&sBhi[r][c4] = make_uint2(h0 | (h1 << 16), h2 | (h3 << 16));
      *(uint2*)&sBlo[r][c4] = make_uint2(l0 | (l1 << 16), l2 | (l3 << 16));
    }
    __syncthreads();
    wmma_block(sAhi, sAlo, sBhi, sBlo, waveM, waveN, lrow, half, acc);
  }

  const int rowBase = blockIdx.y * 128 + waveM;
  const int colBase = blockIdx.x * 128 + waveN;
#pragma unroll
  for (int i = 0; i < 2; ++i) {
    const int r0 = rowBase + i * 16 + half * 8;
    float aadd[8];
#pragma unroll
    for (int v = 0; v < 8; ++v) aadd[v] = avec[r0 + v];
#pragma unroll
    for (int j = 0; j < 4; ++j) {
      const int col  = colBase + j * 16 + lrow;
      const float badd = bvec[col];
      float* op = out + (size_t)r0 * M_COLS + col;
#pragma unroll
      for (int v = 0; v < 8; ++v)
        op[(size_t)v * M_COLS] = acc[i][j][v] + aadd[v] + badd;
    }
  }
}

extern "C" void kernel_launch(void* const* d_in, const int* in_sizes, int n_in,
                              void* d_out, int out_size, void* d_ws, size_t ws_size,
                              hipStream_t stream) {
  (void)in_sizes; (void)n_in; (void)out_size;
  const float* h = (const float*)d_in[0];   // (1, 8192, 1024) fp32
  const float* u = (const float*)d_in[1];   // (1, 1024, 1024) fp32
  const float* w = (const float*)d_in[2];   // (1, 3072) fp32
  float* out  = (float*)d_out;              // (8192, 1024) fp32

  float* avec = (float*)d_ws;               // 8192 floats
  float* bvec = avec + N_ROWS;              // 1024 floats

  const size_t elemsA   = (size_t)N_ROWS * D_DIM;
  const size_t elemsB   = (size_t)M_COLS * D_DIM;
  const size_t vecBytes = (size_t)(N_ROWS + M_COLS) * sizeof(float);
  const size_t needBytes = vecBytes + 2 * elemsA * 2 + 2 * elemsB * 2; // ~36 MB

  if (ws_size >= needBytes) {
    unsigned short* Ahi = (unsigned short*)(bvec + M_COLS);
    unsigned short* Alo = Ahi + elemsA;
    unsigned short* Bhi = Alo + elemsA;
    unsigned short* Blo = Bhi + elemsB;
    convert_kernel<<<dim3(N_ROWS + M_COLS), dim3(256), 0, stream>>>(
        h, u, w, Ahi, Alo, Bhi, Blo, avec, bvec);
    sim_gemm_pre_kernel<<<dim3(M_COLS / 128, N_ROWS / 128), dim3(256), 0, stream>>>(
        Ahi, Alo, Bhi, Blo, avec, bvec, out);
  } else {
    rowdot_kernel<<<dim3((N_ROWS + M_COLS) / 8), dim3(256), 0, stream>>>(h, u, w, avec, bvec);
    sim_gemm_fused_kernel<<<dim3(M_COLS / 128, N_ROWS / 128), dim3(256), 0, stream>>>(
        h, u, w, avec, bvec, out);
  }
}
